// RGN2_Naive_7198365188804
// MI455X (gfx1250) — compile-verified, matrix operand order checked by reading, not run on hardware
//
#include <hip/hip_runtime.h>

// ---------------------------------------------------------------------------
// Peephole LSTM, B=64 L=1024 D=512 H=1024 on MI455X (gfx1250, wave32, WMMA).
//
//  * [W_ih | W_hh] fused -> bf16 Wp (4096 x 1536), row-major (K contiguous).
//  * Input transposed/converted to xT (L,B,D) bf16: step t = contiguous tile.
//  * One persistent kernel, 64 blocks x 128 threads (4 waves), runs all 1024
//    steps with a monotonic-atomic device-wide barrier between steps.
//      block -> 16 hidden columns, wave -> 16 batch rows, 4 gate accumulators.
//  * Per-block weight slice (192 KiB) staged ONCE into LDS in fragment-
//    swizzled order (lane-linear 32B chunks -> conflict-free ds_load_b128);
//    inner loop B-operands come from LDS, only A-operands from global/L2.
//  * c and masked h state live in registers for the whole recurrence; final
//    hx/cx written once at the end. h ping-pongs in bf16 globally (next-step
//    A operand for all blocks).
// ---------------------------------------------------------------------------

typedef __bf16 bf16_t;
typedef __attribute__((ext_vector_type(8)))  __bf16 v8bf;
typedef __attribute__((ext_vector_type(16))) __bf16 v16bf;
typedef __attribute__((ext_vector_type(8)))  float  v8f;

#define B_    64
#define L_    1024
#define D_    512
#define H_    1024
#define KTOT  1536        // D_ + H_
#define G4H   4096        // 4*H_
#define GRID_ 64          // persistent blocks (co-resident on MI455X)
#define NKK   48          // KTOT / 32 k-steps
#define LDS_BYTES (4 * NKK * 1024)   // 4 gates * 48 ksteps * 32 lanes * 32B = 192 KiB

__device__ __forceinline__ float sigm_f(float x) {
  return 1.0f / (1.0f + __expf(-x));
}
__device__ __forceinline__ float tanh_f(float x) {
  return 2.0f * sigm_f(2.0f * x) - 1.0f;      // numerically stable
}

// Global A-operand fragment: two 16B chunks 32B apart
// (VGPR0-3 hold K = kh+{0..7}, VGPR4-7 hold K = kh+16+{0..7}).
__device__ __forceinline__ v16bf load_frag(const bf16_t* p) {
  v8bf lo = *reinterpret_cast<const v8bf*>(p);
  v8bf hi = *reinterpret_cast<const v8bf*>(p + 16);
  return __builtin_shufflevector(lo, hi, 0,1,2,3,4,5,6,7,8,9,10,11,12,13,14,15);
}

// ---------------------------------------------------------------------------
// Prologue kernels
// ---------------------------------------------------------------------------
__global__ void lstm_pack_weights(const float* __restrict__ wih,
                                  const float* __restrict__ whh,
                                  const float* __restrict__ bih,
                                  const float* __restrict__ bhh,
                                  bf16_t* __restrict__ Wp,
                                  float* __restrict__ biasc) {
  long idx = (long)blockIdx.x * blockDim.x + threadIdx.x;
  const long total = (long)G4H * KTOT;
  if (idx < total) {
    int r = (int)(idx / KTOT);
    int k = (int)(idx % KTOT);
    float v = (k < D_) ? wih[(long)r * D_ + k] : whh[(long)r * H_ + (k - D_)];
    Wp[idx] = (bf16_t)v;
  }
  if (idx < G4H) biasc[idx] = bih[idx] + bhh[idx];
}

__global__ void lstm_convert_x(const float* __restrict__ x,
                               bf16_t* __restrict__ xT) {
  long idx = (long)blockIdx.x * blockDim.x + threadIdx.x;
  const long total = (long)L_ * B_ * D_;
  if (idx >= total) return;
  int t = (int)(idx / (B_ * D_));
  int rem = (int)(idx % (B_ * D_));
  int m = rem / D_;
  int k = rem % D_;
  xT[idx] = (bf16_t)x[(long)m * L_ * D_ + (long)t * D_ + k];
}

__global__ void lstm_zero_state(bf16_t* __restrict__ h0,
                                float* __restrict__ hf,
                                float* __restrict__ cf,
                                unsigned int* __restrict__ ctr) {
  int idx = blockIdx.x * blockDim.x + threadIdx.x;
  if (idx < B_ * H_) {
    h0[idx] = (bf16_t)0.0f;
    hf[idx] = 0.0f;
    cf[idx] = 0.0f;
  }
  if (idx == 0) *ctr = 0u;   // reset monotonic barrier every launch (graph-safe)
}

// ---------------------------------------------------------------------------
// Device-wide step barrier: monotonic counter, sense-free.
// ---------------------------------------------------------------------------
__device__ __forceinline__ void grid_barrier(unsigned int* ctr, unsigned int target) {
  __syncthreads();
  if (threadIdx.x == 0) {
    __threadfence();                       // release this block's writes
    atomicAdd(ctr, 1u);
    while (atomicAdd(ctr, 0u) < target) {
      __builtin_amdgcn_s_sleep(1);
    }
  }
  __syncthreads();
  __threadfence();                         // acquire other blocks' writes
}

// ---------------------------------------------------------------------------
// Persistent recurrence kernel: 64 blocks x 128 threads (4 waves).
// ---------------------------------------------------------------------------
__global__ __launch_bounds__(128)
void lstm_persistent(const bf16_t* __restrict__ xT,
                     const bf16_t* __restrict__ Wp,
                     const float*  __restrict__ biasc,
                     bf16_t* __restrict__ h0buf,
                     bf16_t* __restrict__ h1buf,
                     float*  __restrict__ hf,
                     float*  __restrict__ cf,
                     const float* __restrict__ wci,
                     const float* __restrict__ wcf,
                     const float* __restrict__ wco,
                     const unsigned char* __restrict__ mask,
                     float* __restrict__ outs,
                     unsigned int* __restrict__ ctr) {
  extern __shared__ __align__(16) char smem_raw[];
  bf16_t* sW = (bf16_t*)smem_raw;

  const int wave = threadIdx.x >> 5;
  const int lane = threadIdx.x & 31;
  const int m0 = wave * 16;                 // wave's 16 batch rows
  const int n0 = blockIdx.x * 16;           // block's 16 hidden columns
  const int r  = lane & 15;
  const int kh = (lane >> 4) * 8;           // half-wave K sub-offset
  const int j  = n0 + r;                    // this lane's hidden column (C/D layout)
  const int mhi = (lane >> 4) * 8;          // row offset for C/D halves

  // ---- Stage this block's weight slice into LDS, fragment-swizzled. ----
  // Chunk c = ((g*NKK + kk)*32 + lane)*2 + half  (16B each); lane-linear 32B
  // per lane per fragment => conflict-free ds_load_b128 in the hot loop.
  for (int idx = threadIdx.x; idx < 4 * NKK * 64; idx += blockDim.x) {
    const int half   = idx & 1;
    const int lane_s = (idx >> 1) & 31;
    const int gk     = idx >> 6;           // g*NKK + kk
    const int kk     = gk % NKK;
    const int g      = gk / NKK;
    const int rr     = lane_s & 15;
    const int khh    = (lane_s >> 4) * 8 + half * 16;
    const bf16_t* src = Wp + (long)(g * H_ + n0 + rr) * KTOT + kk * 32 + khh;
    *reinterpret_cast<v8bf*>(sW + (long)idx * 8) = *reinterpret_cast<const v8bf*>(src);
  }
  __syncthreads();

  const bf16_t* sw = sW + lane * 16;        // this lane's 32B chunk base

  // Hoisted per-column constants.
  const float bi = biasc[j],          bfc = biasc[H_ + j];
  const float bc = biasc[2 * H_ + j], bo  = biasc[3 * H_ + j];
  const float pci = wci[j], pcf = wcf[j], pco = wco[j];

  // c and masked-h state live in registers for the whole recurrence.
  float creg[8], hreg[8];
#pragma unroll
  for (int v = 0; v < 8; ++v) { creg[v] = 0.0f; hreg[v] = 0.0f; }

  for (int t = 0; t < L_; ++t) {
    const bf16_t* hprev = (t & 1) ? h1buf : h0buf;
    bf16_t*       hnext = (t & 1) ? h0buf : h1buf;

    const bf16_t* arow_x = xT + (long)t * (B_ * D_) + (long)(m0 + r) * D_ + kh;
    const bf16_t* arow_h = hprev + (long)(m0 + r) * H_ + kh;

    v8f acc_i = {}, acc_f = {}, acc_c = {}, acc_o = {};

    // K = [0, 512): x_t contribution (A from global, B from LDS)
#pragma unroll 4
    for (int kk = 0; kk < 16; ++kk) {
      v16bf a  = load_frag(arow_x + kk * 32);
      v16bf b0 = *(const v16bf*)(sw + (0 * NKK + kk) * 512);
      v16bf b1 = *(const v16bf*)(sw + (1 * NKK + kk) * 512);
      v16bf b2 = *(const v16bf*)(sw + (2 * NKK + kk) * 512);
      v16bf b3 = *(const v16bf*)(sw + (3 * NKK + kk) * 512);
      acc_i = __builtin_amdgcn_wmma_f32_16x16x32_bf16(false, a, false, b0, (short)0, acc_i, false, false);
      acc_f = __builtin_amdgcn_wmma_f32_16x16x32_bf16(false, a, false, b1, (short)0, acc_f, false, false);
      acc_c = __builtin_amdgcn_wmma_f32_16x16x32_bf16(false, a, false, b2, (short)0, acc_c, false, false);
      acc_o = __builtin_amdgcn_wmma_f32_16x16x32_bf16(false, a, false, b3, (short)0, acc_o, false, false);
    }
    // K = [512, 1536): h_{t-1} contribution
#pragma unroll 4
    for (int kk = 16; kk < NKK; ++kk) {
      v16bf a  = load_frag(arow_h + (kk - 16) * 32);
      v16bf b0 = *(const v16bf*)(sw + (0 * NKK + kk) * 512);
      v16bf b1 = *(const v16bf*)(sw + (1 * NKK + kk) * 512);
      v16bf b2 = *(const v16bf*)(sw + (2 * NKK + kk) * 512);
      v16bf b3 = *(const v16bf*)(sw + (3 * NKK + kk) * 512);
      acc_i = __builtin_amdgcn_wmma_f32_16x16x32_bf16(false, a, false, b0, (short)0, acc_i, false, false);
      acc_f = __builtin_amdgcn_wmma_f32_16x16x32_bf16(false, a, false, b1, (short)0, acc_f, false, false);
      acc_c = __builtin_amdgcn_wmma_f32_16x16x32_bf16(false, a, false, b2, (short)0, acc_c, false, false);
      acc_o = __builtin_amdgcn_wmma_f32_16x16x32_bf16(false, a, false, b3, (short)0, acc_o, false, false);
    }

    // Gate fusion in registers. Lane owns column j; VGPR v holds row m0+mhi+v.
#pragma unroll
    for (int v = 0; v < 8; ++v) {
      const int m = m0 + mhi + v;
      const long sidx = (long)m * H_ + j;
      const float cprev = creg[v];
      const bool  live  = mask[(long)m * L_ + t] != 0;

      const float ig = sigm_f(acc_i[v] + bi  + pci * cprev);
      const float fg = sigm_f(acc_f[v] + bfc + pcf * cprev);
      const float gg = tanh_f(acc_c[v] + bc);
      const float cy = fg * cprev + ig * gg;
      const float og = sigm_f(acc_o[v] + bo + pco * cy);
      const float hy = og * tanh_f(cy);

      const float hn = live ? hy : hreg[v];
      creg[v] = live ? cy : cprev;
      hreg[v] = hn;

      hnext[sidx] = (bf16_t)hn;
      outs[(long)m * (L_ * H_) + (long)t * H_ + j] = hn;
    }

    grid_barrier(ctr, (unsigned int)(GRID_ * (t + 1)));
  }

  // Final hx / cx written once.
#pragma unroll
  for (int v = 0; v < 8; ++v) {
    const int m = m0 + mhi + v;
    const long sidx = (long)m * H_ + j;
    hf[sidx] = hreg[v];
    cf[sidx] = creg[v];
  }
}

// ---------------------------------------------------------------------------
// Host entry
// ---------------------------------------------------------------------------
extern "C" void kernel_launch(void* const* d_in, const int* in_sizes, int n_in,
                              void* d_out, int out_size, void* d_ws, size_t ws_size,
                              hipStream_t stream) {
  (void)in_sizes; (void)n_in; (void)out_size; (void)ws_size;

  const float*         input = (const float*)d_in[0];         // (B,L,D)
  const unsigned char* mask  = (const unsigned char*)d_in[1]; // (B,L) bool
  const float*         wih   = (const float*)d_in[2];         // (4H,D)
  const float*         whh   = (const float*)d_in[3];         // (4H,H)
  const float*         bih   = (const float*)d_in[4];         // (4H)
  const float*         bhh   = (const float*)d_in[5];         // (4H)
  const float*         wci   = (const float*)d_in[6];         // (H)
  const float*         wcf   = (const float*)d_in[7];         // (H)
  const float*         wco   = (const float*)d_in[8];         // (H)
  float* out = (float*)d_out;  // outs (B,L,H) | hx (B,H) | cx (B,H)

  char* ws = (char*)d_ws;
  auto take = [&](size_t bytes) -> void* {
    void* p = (void*)ws;
    ws += (bytes + 255) & ~(size_t)255;
    return p;
  };
  bf16_t* Wp    = (bf16_t*)take((size_t)G4H * KTOT * sizeof(bf16_t));   // 12.6 MB
  float*  biasc = (float*) take((size_t)G4H * sizeof(float));
  bf16_t* xT    = (bf16_t*)take((size_t)L_ * B_ * D_ * sizeof(bf16_t)); // 67 MB
  bf16_t* h0    = (bf16_t*)take((size_t)B_ * H_ * sizeof(bf16_t));
  bf16_t* h1    = (bf16_t*)take((size_t)B_ * H_ * sizeof(bf16_t));
  float*  hf    = (float*) take((size_t)B_ * H_ * sizeof(float));
  float*  cf    = (float*) take((size_t)B_ * H_ * sizeof(float));
  unsigned int* ctr = (unsigned int*)take(256);

  // Allow 192 KiB dynamic LDS (hardware WGP LDS is 320 KiB). Host-side call,
  // not a stream operation -> graph-capture safe and deterministic.
  static_assert(LDS_BYTES <= 320 * 1024, "LDS slice must fit in WGP LDS");
  hipFuncSetAttribute(reinterpret_cast<const void*>(lstm_persistent),
                      hipFuncAttributeMaxDynamicSharedMemorySize, LDS_BYTES);

  {
    const long total = (long)G4H * KTOT;
    lstm_pack_weights<<<(unsigned)((total + 255) / 256), 256, 0, stream>>>(
        wih, whh, bih, bhh, Wp, biasc);
  }
  {
    const long total = (long)L_ * B_ * D_;
    lstm_convert_x<<<(unsigned)((total + 255) / 256), 256, 0, stream>>>(input, xT);
  }
  lstm_zero_state<<<(B_ * H_ + 255) / 256, 256, 0, stream>>>(h0, hf, cf, ctr);

  lstm_persistent<<<GRID_, 128, LDS_BYTES, stream>>>(
      xT, Wp, biasc, h0, h1, hf, cf, wci, wcf, wco, mask, out, ctr);

  // Final hx, cx
  hipMemcpyAsync(out + (size_t)B_ * L_ * H_, hf,
                 (size_t)B_ * H_ * sizeof(float), hipMemcpyDeviceToDevice, stream);
  hipMemcpyAsync(out + (size_t)B_ * L_ * H_ + (size_t)B_ * H_, cf,
                 (size_t)B_ * H_ * sizeof(float), hipMemcpyDeviceToDevice, stream);
}
